// TPTransformerParallelBlock_70866960384632
// MI455X (gfx1250) — matrix operand order repaired and checked by reading, not compiled, not run on hardware
//
#include <hip/hip_runtime.h>

// ---------------------------------------------------------------------------
// TPTransformerParallelBlock for MI455X (gfx1250, wave32, WMMA).
// B=1, S=4096, D=1024, H=16, Dh=64, F=4096. fp32 in/out, bf16 WMMA compute.
// GEMMs: double-buffered LDS staged via async global->LDS (ASYNCcnt),
// pre-transposed bf16 weights so all staging is contiguous b128 copies.
// ---------------------------------------------------------------------------

typedef unsigned short u16;
typedef unsigned int   u32;
typedef __attribute__((ext_vector_type(4)))  u32     u32x4;
typedef __attribute__((ext_vector_type(8)))  float   v8f;
typedef __attribute__((ext_vector_type(16))) __bf16  v16bf;

union Frag {            // 32 bytes: one 16x32 (A) or 32x16 (B) bf16 operand slice
    v16bf v;
    u32x4 q[2];
};

#define S_LEN 4096
#define D_DIM 1024
#define H_N   16
#define DH    64
#define F_DIM 4096

__device__ __forceinline__ u16 f2bf(float f) {
    u32 u = __float_as_uint(f);
    u32 r = u + 0x7FFFu + ((u >> 16) & 1u);   // round-to-nearest-even
    return (u16)(r >> 16);
}

__device__ __forceinline__ float gelu_tanh(float x) {
    float x3 = x * x * x;
    float t  = tanhf(0.7978845608028654f * (x + 0.044715f * x3));
    return 0.5f * x * (1.0f + t);
}

// async 16-byte global->LDS copy (per lane), tracked by ASYNCcnt
__device__ __forceinline__ void async_copy_b128(u32 lds_off, const void* gaddr) {
    asm volatile("global_load_async_to_lds_b128 %0, %1, off"
                 :: "v"(lds_off), "v"(gaddr) : "memory");
}

// ------------------------------- elementwise -------------------------------

__global__ void __launch_bounds__(256)
cast_bf16_kernel(const float* __restrict__ in, u16* __restrict__ out, int n) {
    int i = blockIdx.x * 256 + threadIdx.x;
    if (i < n) out[i] = f2bf(in[i]);
}

// cast + transpose: in [K][N] fp32 -> out [N][K] bf16
__global__ void __launch_bounds__(256)
cast_transpose_kernel(const float* __restrict__ in, u16* __restrict__ out,
                      int K, int N) {
    int idx = blockIdx.x * 256 + threadIdx.x;
    if (idx < K * N) {
        int k = idx / N, n = idx - k * N;
        out[(size_t)n * K + k] = f2bf(in[idx]);
    }
}

__global__ void __launch_bounds__(256)
add3_kernel(const float* __restrict__ x, const float* __restrict__ a,
            const float* __restrict__ b, float* __restrict__ out, int n) {
    int i = blockIdx.x * 256 + threadIdx.x;
    if (i < n) out[i] = x[i] + a[i] + b[i];
}

// ------------------------- pre-norm LayerNorm (D=1024) ---------------------

__global__ void __launch_bounds__(256)
ln_cast_kernel(const float* __restrict__ x, const float* __restrict__ scale,
               const float* __restrict__ bias, u16* __restrict__ out) {
    const int s = blockIdx.x;
    __shared__ float redS[8], redQ[8];
    const float* row = x + (size_t)s * D_DIM;
    float sum = 0.f, sq = 0.f;
    for (int d = threadIdx.x; d < D_DIM; d += 256) {
        float v = row[d]; sum += v; sq += v * v;
    }
    #pragma unroll
    for (int o = 16; o >= 1; o >>= 1) {
        sum += __shfl_xor(sum, o, 32);
        sq  += __shfl_xor(sq,  o, 32);
    }
    if ((threadIdx.x & 31) == 0) {
        redS[threadIdx.x >> 5] = sum;
        redQ[threadIdx.x >> 5] = sq;
    }
    __syncthreads();
    float ts = 0.f, tq = 0.f;
    #pragma unroll
    for (int i = 0; i < 8; ++i) { ts += redS[i]; tq += redQ[i]; }
    const float mean = ts * (1.0f / D_DIM);
    const float var  = tq * (1.0f / D_DIM) - mean * mean;
    const float inv  = rsqrtf(var + 1e-6f);
    for (int d = threadIdx.x; d < D_DIM; d += 256) {
        float v = (row[d] - mean) * inv * scale[d] + bias[d];
        out[(size_t)s * D_DIM + d] = f2bf(v);
    }
}

// ------------------ per-head qk LayerNorm + relayout [H][S][64] ------------

__global__ void __launch_bounds__(32)
head_ln_kernel(const float* __restrict__ qf, const float* __restrict__ bias,
               const float* __restrict__ scale, u16* __restrict__ out, int doLN) {
    const int s = blockIdx.x, h = blockIdx.y, lane = threadIdx.x;
    const float* row = qf + (size_t)s * (H_N * DH) + h * DH;
    float v0 = row[lane]      + bias[h * DH + lane];
    float v1 = row[lane + 32] + bias[h * DH + lane + 32];
    if (doLN) {
        float sum = v0 + v1, sq = v0 * v0 + v1 * v1;
        #pragma unroll
        for (int o = 16; o >= 1; o >>= 1) {
            sum += __shfl_xor(sum, o, 32);
            sq  += __shfl_xor(sq,  o, 32);
        }
        float mean = sum * (1.0f / DH);
        float var  = sq * (1.0f / DH) - mean * mean;
        float inv  = rsqrtf(var + 1e-6f);
        v0 = (v0 - mean) * inv * scale[lane];
        v1 = (v1 - mean) * inv * scale[lane + 32];
    }
    size_t base = (((size_t)h * S_LEN) + s) * DH;
    out[base + lane]      = f2bf(v0);
    out[base + lane + 32] = f2bf(v1);
}

// ------------------- V: bias + cast + transpose to [H][64][S] --------------

__global__ void __launch_bounds__(256)
v_relayout_kernel(const float* __restrict__ vf, const float* __restrict__ bias,
                  u16* __restrict__ outT) {
    int idx = blockIdx.x * 256 + threadIdx.x;       // over S*1024
    int s = idx >> 10, c = idx & 1023;
    int h = c >> 6, d = c & 63;
    float v = vf[idx] + bias[c];
    outT[(((size_t)h * DH) + d) * S_LEN + s] = f2bf(v);
}

// ------------------- bf16 WMMA GEMM: C = A[M,K] * Bt[N,K]^T ----------------
// 128x128 block tile, TK=32, 8 waves in a 4x2 grid; each wave owns 32x64.
// Both tiles staged [row][k] via async global->LDS b128 with double buffering.

#define TM 128
#define TN 128
#define TK 32

__global__ void __launch_bounds__(256)
gemm_bf16_wmma(const u16* __restrict__ A, const u16* __restrict__ Bt,
               const float* __restrict__ bias,
               float* __restrict__ outF, u16* __restrict__ outB,
               int M, int N, int K, int applyGelu) {
    __shared__ __align__(16) u16 lA[2][TM * TK];     // 2 x 8 KB
    __shared__ __align__(16) u16 lB[2][TN * TK];     // 2 x 8 KB  ([n][k])
    const int tid  = threadIdx.x;
    const int wave = tid >> 5;
    const int lane = tid & 31;
    const int grp  = lane >> 4;       // half-wave: 0 or 1
    const int lm   = lane & 15;
    const int wy   = wave >> 1;       // 0..3 -> 32-row strip
    const int wx   = wave & 1;        // 0..1 -> 64-col strip
    const int bm   = blockIdx.y * TM;
    const int bn   = blockIdx.x * TN;

    v8f acc[2][4] = {};

    // issue one k-tile's staging (4 async b128 per thread-slot -> 4/wave)
    auto stage = [&](int buf, int k0) {
        #pragma unroll
        for (int i = 0; i < 2; ++i) {
            int idx = tid + i * 256;          // 0..511
            int r = idx >> 2;                 // row 0..127
            int c = (idx & 3) * 8;            // k-col 0,8,16,24
            async_copy_b128((u32)(size_t)&lA[buf][r * TK + c],
                            &A[(size_t)(bm + r) * K + k0 + c]);
            async_copy_b128((u32)(size_t)&lB[buf][r * TK + c],
                            &Bt[(size_t)(bn + r) * K + k0 + c]);
        }
    };

    const int nk = K / TK;
    stage(0, 0);
    for (int t = 0; t < nk; ++t) {
        const int cur = t & 1;
        if (t + 1 < nk) {
            stage(cur ^ 1, (t + 1) * TK);
            asm volatile("s_wait_asynccnt 0x4" ::: "memory");  // tile t done
        } else {
            asm volatile("s_wait_asynccnt 0x0" ::: "memory");
        }
        __syncthreads();

        Frag aF[2], bF[4];
        #pragma unroll
        for (int im = 0; im < 2; ++im) {
            int row = wy * 32 + im * 16 + lm;
            aF[im].q[0] = *(const u32x4*)&lA[cur][row * TK + grp * 8];
            aF[im].q[1] = *(const u32x4*)&lA[cur][row * TK + 16 + grp * 8];
        }
        #pragma unroll
        for (int in = 0; in < 4; ++in) {
            int col = wx * 64 + in * 16 + lm;
            bF[in].q[0] = *(const u32x4*)&lB[cur][col * TK + grp * 8];
            bF[in].q[1] = *(const u32x4*)&lB[cur][col * TK + 16 + grp * 8];
        }
        #pragma unroll
        for (int im = 0; im < 2; ++im)
            #pragma unroll
            for (int in = 0; in < 4; ++in)
                acc[im][in] = __builtin_amdgcn_wmma_f32_16x16x32_bf16(
                    false, aF[im].v, false, bF[in].v, (short)0, acc[im][in],
                    false, false);
        __syncthreads();   // protect cur buffer before it is restaged
    }

    // ---- epilogue: C layout VGPR i -> m = grp*8 + i, n = lm ----
    #pragma unroll
    for (int im = 0; im < 2; ++im) {
        #pragma unroll
        for (int in = 0; in < 4; ++in) {
            int n = bn + wx * 64 + in * 16 + lm;
            float bv = bias ? bias[n] : 0.0f;
            #pragma unroll
            for (int i = 0; i < 8; ++i) {
                int m = bm + wy * 32 + im * 16 + grp * 8 + i;
                float v = acc[im][in][i] + bv;
                if (applyGelu) v = gelu_tanh(v);
                if (outF) outF[(size_t)m * N + n] = v;
                if (outB) outB[(size_t)m * N + n] = f2bf(v);
            }
        }
    }
}

// --------------------- flash attention, one wave / 16 q rows ---------------
// qb,kb: [H][S][64] bf16.  vT: [H][64][S] bf16.  o: [S][H*64] bf16.

__global__ void __launch_bounds__(32)
flash_attn_kernel(const u16* __restrict__ qb, const u16* __restrict__ kb,
                  const u16* __restrict__ vT, u16* __restrict__ o) {
    const int h  = blockIdx.x;
    const int qt = blockIdx.y;            // 16-row query tile index
    const int lane = threadIdx.x;
    const int grp  = lane >> 4;
    const int lm   = lane & 15;
    __shared__ __align__(16) u16 lP[16 * 16];

    // Q fragments: A 16x64 = two K=32 slices
    const u16* qrow = qb + (((size_t)h * S_LEN) + qt * 16 + lm) * DH;
    Frag qf0, qf1;
    qf0.q[0] = *(const u32x4*)&qrow[grp * 8];
    qf0.q[1] = *(const u32x4*)&qrow[16 + grp * 8];
    qf1.q[0] = *(const u32x4*)&qrow[32 + grp * 8];
    qf1.q[1] = *(const u32x4*)&qrow[48 + grp * 8];

    float mrun[8], lrun[8];
    #pragma unroll
    for (int i = 0; i < 8; ++i) { mrun[i] = -1e30f; lrun[i] = 0.0f; }
    v8f oacc[4] = {};
    const u32x4 zero4 = {0u, 0u, 0u, 0u};

    for (int kt = 0; kt <= qt; ++kt) {
        // K^T fragments: B[64][16], n = key = kt*16+lm
        const u16* krow = kb + (((size_t)h * S_LEN) + kt * 16 + lm) * DH;
        Frag kf0, kf1;
        kf0.q[0] = *(const u32x4*)&krow[grp * 8];
        kf0.q[1] = *(const u32x4*)&krow[16 + grp * 8];
        kf1.q[0] = *(const u32x4*)&krow[32 + grp * 8];
        kf1.q[1] = *(const u32x4*)&krow[48 + grp * 8];

        v8f sc = {};
        sc = __builtin_amdgcn_wmma_f32_16x16x32_bf16(false, qf0.v, false, kf0.v,
                                                     (short)0, sc, false, false);
        sc = __builtin_amdgcn_wmma_f32_16x16x32_bf16(false, qf1.v, false, kf1.v,
                                                     (short)0, sc, false, false);

        // scale + causal mask + online softmax (row m = grp*8+i, col n = lm)
        #pragma unroll
        for (int i = 0; i < 8; ++i) {
            float v = sc[i] * 0.125f;                       // 1/sqrt(64)
            int m = grp * 8 + i;
            if (kt == qt && lm > m) v = -1e30f;
            float rm = v;
            #pragma unroll
            for (int ox = 8; ox >= 1; ox >>= 1)
                rm = fmaxf(rm, __shfl_xor(rm, ox, 32));     // within 16-lane group
            float mn    = fmaxf(mrun[i], rm);
            float alpha = __expf(mrun[i] - mn);
            float p     = __expf(v - mn);
            float rs = p;
            #pragma unroll
            for (int ox = 8; ox >= 1; ox >>= 1)
                rs += __shfl_xor(rs, ox, 32);
            lrun[i] = lrun[i] * alpha + rs;
            mrun[i] = mn;
            #pragma unroll
            for (int t = 0; t < 4; ++t) oacc[t][i] *= alpha;
            sc[i] = p;
        }

        // P -> LDS (bf16), re-fragment as A operand (K 16..31 zero-padded)
        #pragma unroll
        for (int i = 0; i < 8; ++i)
            lP[(grp * 8 + i) * 16 + lm] = f2bf(sc[i]);
        __syncthreads();
        Frag pf;
        pf.q[0] = *(const u32x4*)&lP[lm * 16 + grp * 8];
        pf.q[1] = zero4;

        // V fragments from vT: key idx = grp*8 + s for s < 8, elements 8..15 zero
        #pragma unroll
        for (int t = 0; t < 4; ++t) {
            Frag vf;
            {
                const u16* vrow = vT + (((size_t)h * DH) + t * 16 + lm) * S_LEN + kt * 16;
                vf.q[0] = *(const u32x4*)&vrow[grp * 8];
                vf.q[1] = zero4;
            }
            oacc[t] = __builtin_amdgcn_wmma_f32_16x16x32_bf16(
                false, pf.v, false, vf.v, (short)0, oacc[t], false, false);
        }
        __syncthreads();
    }

    // normalize & store: o[qt*16+m][h*64 + t*16 + lm]
    #pragma unroll
    for (int i = 0; i < 8; ++i) {
        float invl = 1.0f / lrun[i];
        int m = qt * 16 + grp * 8 + i;
        #pragma unroll
        for (int t = 0; t < 4; ++t)
            o[(size_t)m * (H_N * DH) + h * DH + t * 16 + lm] = f2bf(oacc[t][i] * invl);
    }
}

// ---------------------------------------------------------------------------

extern "C" void kernel_launch(void* const* d_in, const int* in_sizes, int n_in,
                              void* d_out, int out_size, void* d_ws, size_t ws_size,
                              hipStream_t stream) {
    (void)in_sizes; (void)n_in; (void)out_size; (void)ws_size;
    const float* x        = (const float*)d_in[0];
    // d_in[1] = mask (causal; handled analytically)
    const float* ln_scale = (const float*)d_in[2];
    const float* ln_bias  = (const float*)d_in[3];
    const float* w_in     = (const float*)d_in[4];
    const float* b_in     = (const float*)d_in[5];
    const float* wq       = (const float*)d_in[6];
    const float* bq       = (const float*)d_in[7];
    const float* wk       = (const float*)d_in[8];
    const float* bk       = (const float*)d_in[9];
    const float* wv       = (const float*)d_in[10];
    const float* bv       = (const float*)d_in[11];
    const float* qn_scale = (const float*)d_in[12];
    const float* kn_scale = (const float*)d_in[13];
    const float* w_mo     = (const float*)d_in[14];
    const float* b_mo     = (const float*)d_in[15];
    const float* w_ao     = (const float*)d_in[16];
    const float* b_ao     = (const float*)d_in[17];
    float* out = (float*)d_out;

    // ---- workspace layout (bytes) ----
    char* ws = (char*)d_ws;
    size_t off = 0;
    auto alloc = [&](size_t bytes) { void* p = ws + off; off += (bytes + 255) & ~(size_t)255; return p; };
    u16*   xn_b    = (u16*)alloc((size_t)S_LEN * D_DIM * 2);
    u16*   w_in_t  = (u16*)alloc((size_t)D_DIM * F_DIM * 2);        // [F][D]
    u16*   wq_t    = (u16*)alloc((size_t)D_DIM * H_N * DH * 2);     // [1024][D]
    u16*   wk_t    = (u16*)alloc((size_t)D_DIM * H_N * DH * 2);
    u16*   wv_t    = (u16*)alloc((size_t)D_DIM * H_N * DH * 2);
    u16*   w_mo_t  = (u16*)alloc((size_t)F_DIM * D_DIM * 2);        // [D][F]
    u16*   w_ao_t  = (u16*)alloc((size_t)H_N * DH * D_DIM * 2);     // [D][1024]
    u16*   h_b     = (u16*)alloc((size_t)S_LEN * F_DIM * 2);
    float* proj_f  = (float*)alloc((size_t)S_LEN * H_N * DH * 4);   // q/k/v staging
    u16*   q_b     = (u16*)alloc((size_t)H_N * S_LEN * DH * 2);
    u16*   k_b     = (u16*)alloc((size_t)H_N * S_LEN * DH * 2);
    u16*   vT_b    = (u16*)alloc((size_t)H_N * DH * S_LEN * 2);
    u16*   o_b     = (u16*)alloc((size_t)S_LEN * H_N * DH * 2);
    float* mlp_f   = (float*)alloc((size_t)S_LEN * D_DIM * 4);
    float* attn_f  = (float*)alloc((size_t)S_LEN * D_DIM * 4);

    const int NSD = S_LEN * D_DIM;

    // 1. pre-norm LayerNorm -> bf16
    ln_cast_kernel<<<S_LEN, 256, 0, stream>>>(x, ln_scale, ln_bias, xn_b);

    // 2. weight casts with transpose (B operand stored [N][K])
    cast_transpose_kernel<<<(D_DIM * F_DIM) / 256, 256, 0, stream>>>(w_in, w_in_t, D_DIM, F_DIM);
    cast_transpose_kernel<<<(D_DIM * H_N * DH) / 256, 256, 0, stream>>>(wq, wq_t, D_DIM, H_N * DH);
    cast_transpose_kernel<<<(D_DIM * H_N * DH) / 256, 256, 0, stream>>>(wk, wk_t, D_DIM, H_N * DH);
    cast_transpose_kernel<<<(D_DIM * H_N * DH) / 256, 256, 0, stream>>>(wv, wv_t, D_DIM, H_N * DH);
    cast_transpose_kernel<<<(F_DIM * D_DIM) / 256, 256, 0, stream>>>(w_mo, w_mo_t, F_DIM, D_DIM);
    cast_transpose_kernel<<<(H_N * DH * D_DIM) / 256, 256, 0, stream>>>(w_ao, w_ao_t, H_N * DH, D_DIM);

    // 3. h = gelu(xn @ w_in + b_in)  [S,F] -> bf16
    gemm_bf16_wmma<<<dim3(F_DIM / TN, S_LEN / TM), 256, 0, stream>>>(
        xn_b, w_in_t, b_in, nullptr, h_b, S_LEN, F_DIM, D_DIM, 1);

    // 4. q/k/v projections + per-head norm + relayout
    gemm_bf16_wmma<<<dim3((H_N * DH) / TN, S_LEN / TM), 256, 0, stream>>>(
        xn_b, wq_t, nullptr, proj_f, nullptr, S_LEN, H_N * DH, D_DIM, 0);
    head_ln_kernel<<<dim3(S_LEN, H_N), 32, 0, stream>>>(proj_f, bq, qn_scale, q_b, 1);

    gemm_bf16_wmma<<<dim3((H_N * DH) / TN, S_LEN / TM), 256, 0, stream>>>(
        xn_b, wk_t, nullptr, proj_f, nullptr, S_LEN, H_N * DH, D_DIM, 0);
    head_ln_kernel<<<dim3(S_LEN, H_N), 32, 0, stream>>>(proj_f, bk, kn_scale, k_b, 1);

    gemm_bf16_wmma<<<dim3((H_N * DH) / TN, S_LEN / TM), 256, 0, stream>>>(
        xn_b, wv_t, nullptr, proj_f, nullptr, S_LEN, H_N * DH, D_DIM, 0);
    v_relayout_kernel<<<NSD / 256, 256, 0, stream>>>(proj_f, bv, vT_b);

    // 5. causal flash attention -> o_b [S, H*64] bf16
    flash_attn_kernel<<<dim3(H_N, S_LEN / 16), 32, 0, stream>>>(q_b, k_b, vT_b, o_b);

    // 6. output projections (fp32)
    gemm_bf16_wmma<<<dim3(D_DIM / TN, S_LEN / TM), 256, 0, stream>>>(
        h_b, w_mo_t, b_mo, mlp_f, nullptr, S_LEN, D_DIM, F_DIM, 0);
    gemm_bf16_wmma<<<dim3(D_DIM / TN, S_LEN / TM), 256, 0, stream>>>(
        o_b, w_ao_t, b_ao, attn_f, nullptr, S_LEN, D_DIM, H_N * DH, 0);

    // 7. residual add
    add3_kernel<<<NSD / 256, 256, 0, stream>>>(x, mlp_f, attn_f, out, NSD);
}
